// HyperPhS_76879914598441
// MI455X (gfx1250) — matrix operand order repaired and checked by reading, not compiled, not run on hardware
//
#include <hip/hip_runtime.h>
#include <hip/hip_bf16.h>

#define N_NODES  51200
#define N_EDGES  204800
#define N_HNNZ   102400
#define N_HEDGES 10240
#define BATCH    1024

typedef __attribute__((ext_vector_type(16))) __bf16 v16bf;
typedef __attribute__((ext_vector_type(8)))  float  v8f;

// ---------------------------------------------------------------------------
// WMMA GEMM: C[M,N] = act(A[M,K] @ W[N,K]^T + bias)
// BM=128, BN=64, BK=32; 256 threads = 8 waves; wave wv owns rows [16*wv,16*wv+16)
// and all 64 cols (4 accumulators, 4 WMMAs per k-step reusing one A fragment).
// Double-buffered LDS: next tile's global loads overlap current tile's WMMAs;
// one barrier per k-step.
// ---------------------------------------------------------------------------
#define BM 128
#define BN 64
#define BK 32
#define LDS_STRIDE (BK + 4)   // 36 bf16 = 72B row stride (8B aligned, conflict-free)

__global__ __launch_bounds__(256) void wmma_gemm_kernel(
    const float* __restrict__ A, int lda,
    const float* __restrict__ W,           // [N,K] row-major
    const float* __restrict__ bias,        // may be null
    float* __restrict__ C, int ldc,
    int M, int N, int K, int act)
{
    __shared__ __bf16 As[2][BM][LDS_STRIDE];
    __shared__ __bf16 Bs[2][BN][LDS_STRIDE];

    const int tid  = threadIdx.x;
    const int lane = tid & 31;
    const int wv   = tid >> 5;            // 0..7 -> M tile

    const int block_m = blockIdx.x * BM;
    const int block_n = blockIdx.y * BN;

    v8f acc[4] = {v8f{}, v8f{}, v8f{}, v8f{}};

    const int fm  = lane & 15;            // M (or N) index within 16
    const int hi  = lane >> 4;            // 0 or 1
    const int kbA = hi << 3;              // A k-base: 0 / 8
    const int kbB = hi << 4;              // B k-base: 0 / 16

    // cooperative-load mapping: A 128x32 -> 16 floats/thread, B 64x32 -> 8/thread
    const int ar = tid >> 1;              // 0..127
    const int ac = (tid & 1) * 16;        // 0 or 16
    const int br = tid >> 2;              // 0..63
    const int bc = (tid & 3) * 8;         // 0,8,16,24

    const bool fullM = (block_m + BM) <= M;
    const bool fullN = (block_n + BN) <= N;

    // global tile -> registers (branch-free in both paths)
    auto load_regs = [&](int k0, float* ra, float* rb) {
        const bool fullK = (k0 + BK) <= K;
        if (fullM && fullN && fullK) {
            const float* ap = A + (size_t)(block_m + ar) * lda + k0 + ac;
            const float* wp = W + (size_t)(block_n + br) * K + k0 + bc;
#pragma unroll
            for (int j = 0; j < 16; ++j) ra[j] = ap[j];
#pragma unroll
            for (int j = 0; j < 8; ++j)  rb[j] = wp[j];
        } else {
            const int gm = block_m + ar;
            const int gn = block_n + br;
            const int cm = gm < M ? gm : (M - 1);
            const int cn = gn < N ? gn : (N - 1);
#pragma unroll
            for (int j = 0; j < 16; ++j) {
                const int gk = k0 + ac + j;
                const int ck = gk < K ? gk : (K - 1);
                float v = A[(size_t)cm * lda + ck];
                ra[j] = (gm < M && gk < K) ? v : 0.0f;
            }
#pragma unroll
            for (int j = 0; j < 8; ++j) {
                const int gk = k0 + bc + j;
                const int ck = gk < K ? gk : (K - 1);
                float v = W[(size_t)cn * K + ck];
                rb[j] = (gn < N && gk < K) ? v : 0.0f;
            }
        }
    };
    auto store_tile = [&](int bsel, const float* ra, const float* rb) {
#pragma unroll
        for (int j = 0; j < 16; ++j) As[bsel][ar][ac + j] = (__bf16)ra[j];
#pragma unroll
        for (int j = 0; j < 8; ++j)  Bs[bsel][br][bc + j] = (__bf16)rb[j];
    };

    float ra[16], rb[8];
    load_regs(0, ra, rb);
    store_tile(0, ra, rb);
    __syncthreads();

    const int nk = (K + BK - 1) / BK;
    int buf = 0;
    for (int it = 0; it < nk; ++it) {
        const bool hasNext = (it + 1) < nk;
        if (hasNext) load_regs((it + 1) * BK, ra, rb);   // overlaps with WMMAs below

        // fragment assembly per ISA 7.12.2 16-bit layouts
        v16bf a;
#pragma unroll
        for (int e = 0; e < 16; ++e) {
            const int ka = ((e < 8) ? e : (e + 8)) + kbA;  // A: 16x32
            a[e] = As[buf][wv * 16 + fm][ka];
        }
#pragma unroll
        for (int t4 = 0; t4 < 4; ++t4) {
            v16bf b;
#pragma unroll
            for (int e = 0; e < 16; ++e) {
                const int kb = ((e < 8) ? e : (e + 8)) + kbA;  // B: 32x16
                b[e] = Bs[buf][t4 * 16 + fm][kb];
            }
            acc[t4] = __builtin_amdgcn_wmma_f32_16x16x32_bf16(
                false, a, false, b, (short)0, acc[t4], false, false);
        }

        if (hasNext) store_tile(buf ^ 1, ra, rb);
        __syncthreads();       // one barrier per k-step (ping-pong)
        buf ^= 1;
    }

    // epilogue: D element r -> row = r + 8*hi, col = lane&15
    const int crow = block_m + wv * 16 + (hi << 3);
#pragma unroll
    for (int t4 = 0; t4 < 4; ++t4) {
        const int col = block_n + t4 * 16 + fm;
        if (col < N) {
            const float bv = bias ? bias[col] : 0.0f;
#pragma unroll
            for (int r = 0; r < 8; ++r) {
                const int row = crow + r;
                if (row < M) {
                    float v = acc[t4][r] + bv;
                    if (act) v = fmaxf(v, 0.0f);
                    C[(size_t)row * ldc + col] = v;
                }
            }
        }
    }
}

// ---------------------------------------------------------------------------
// Elementwise / scatter helpers
// ---------------------------------------------------------------------------
__global__ void zero_kernel(float* p, int n) {
    int i = blockIdx.x * blockDim.x + threadIdx.x;
    if (i < n) p[i] = 0.0f;
}

__global__ void add_inplace_kernel(float* dst, const float* src, int n) {
    int i = blockIdx.x * blockDim.x + threadIdx.x;
    if (i < n) dst[i] += src[i];
}

// dst[idxD[i], f] += src[idxS[i], f]
__global__ void scatter_add_kernel(float* __restrict__ dst, const int* __restrict__ idxD,
                                   const float* __restrict__ src, const int* __restrict__ idxS,
                                   int nitems, int F) {
    int gid = blockIdx.x * blockDim.x + threadIdx.x;
    if (gid >= nitems * F) return;
    int i = gid / F;
    int f = gid - i * F;
    atomicAdd(&dst[(size_t)idxD[i] * F + f], src[(size_t)idxS[i] * F + f]);
}

__global__ void count_kernel(float* cnt, const int* idx, int n) {
    int i = blockIdx.x * blockDim.x + threadIdx.x;
    if (i < n) atomicAdd(&cnt[idx[i]], 1.0f);
}

__global__ void binv_kernel(float* c, int n) {   // 1 / max(c, 1)
    int i = blockIdx.x * blockDim.x + threadIdx.x;
    if (i < n) c[i] = 1.0f / fmaxf(c[i], 1.0f);
}

__global__ void dinv_kernel(float* c, int n) {   // c>0 ? 1/c : 0
    int i = blockIdx.x * blockDim.x + threadIdx.x;
    if (i < n) c[i] = (c[i] > 0.0f) ? (1.0f / c[i]) : 0.0f;
}

__global__ void scale_rows_kernel(float* buf, const float* s, int rows, int F) {
    int gid = blockIdx.x * blockDim.x + threadIdx.x;
    if (gid >= rows * F) return;
    int r = gid / F;
    buf[gid] *= s[r];
}

__global__ void scale_bias_rows_kernel(float* buf, const float* s, const float* bias,
                                       int rows, int F) {
    int gid = blockIdx.x * blockDim.x + threadIdx.x;
    if (gid >= rows * F) return;
    int r = gid / F;
    int f = gid - r * F;
    buf[gid] = buf[gid] * s[r] + bias[f];
}

__global__ void copy_strided_kernel(float* dst, int dstride, const float* src, int sstride,
                                    int rows, int cols) {
    int gid = blockIdx.x * blockDim.x + threadIdx.x;
    if (gid >= rows * cols) return;
    int r = gid / cols;
    int c = gid - r * cols;
    dst[(size_t)r * dstride + c] = src[(size_t)r * sstride + c];
}

// sorted batch ids -> start offsets per graph
__global__ void starts_kernel(int* starts, const int* batch, int n, int nb) {
    int i = blockIdx.x * blockDim.x + threadIdx.x;
    if (i == 0) { starts[0] = 0; starts[nb] = n; }
    if (i > 0 && i < n && batch[i] != batch[i - 1]) starts[batch[i]] = i;
}

// pooled[g] = [max | mean] over nodes of graph g
__global__ void pool_kernel(const float* __restrict__ h, int F, const int* __restrict__ starts,
                            float* __restrict__ out) {
    int g = blockIdx.x;
    int f = blockIdx.y * blockDim.x + threadIdx.x;
    if (f >= F) return;
    int s = starts[g], e = starts[g + 1];
    float mx = -3.4e38f, sm = 0.0f;
    for (int i = s; i < e; ++i) {
        float v = h[(size_t)i * F + f];
        mx = fmaxf(mx, v);
        sm += v;
    }
    out[(size_t)g * (2 * F) + f]     = mx;
    out[(size_t)g * (2 * F) + F + f] = sm / (float)(e - s);
}

// ---------------------------------------------------------------------------
// 4-token / 4-head attention. q,k,v: [B*4, 2048] (token-major, head slice h*512)
// att out: [B][4 heads][4 tokens][512]
// ---------------------------------------------------------------------------
__global__ __launch_bounds__(256) void attn_kernel(const float* __restrict__ q,
                                                   const float* __restrict__ k,
                                                   const float* __restrict__ v,
                                                   float* __restrict__ att) {
    const int b = blockIdx.x >> 2;
    const int h = blockIdx.x & 3;
    const float* qb = q + (size_t)b * 4 * 2048 + h * 512;
    const float* kb = k + (size_t)b * 4 * 2048 + h * 512;
    const float* vb = v + (size_t)b * 4 * 2048 + h * 512;

    __shared__ float red[256];
    __shared__ float s[16];
    __shared__ float dist[16];

    const int t   = threadIdx.x;
    const int p   = t >> 4;     // score pair 0..15
    const int sub = t & 15;
    const int n   = p >> 2, m = p & 3;

    float partial = 0.0f;
    for (int d = sub; d < 512; d += 16)
        partial += qb[n * 2048 + d] * kb[m * 2048 + d];
    red[t] = partial;
    __syncthreads();

    if (sub == 0) {
        float acc = 0.0f;
        for (int j = 0; j < 16; ++j) acc += red[p * 16 + j];
        s[p] = acc * 0.044194173824159216f;  // 1/sqrt(512)
    }
    __syncthreads();

    if (t < 4) {
        float mx = -3.4e38f;
        for (int mm = 0; mm < 4; ++mm) mx = fmaxf(mx, s[t * 4 + mm]);
        float e[4], sum = 0.0f;
        for (int mm = 0; mm < 4; ++mm) { e[mm] = __expf(s[t * 4 + mm] - mx); sum += e[mm]; }
        for (int mm = 0; mm < 4; ++mm) dist[t * 4 + mm] = e[mm] / sum;
    }
    __syncthreads();

    float* ob = att + (size_t)b * 4 * 4 * 512 + (size_t)h * 4 * 512;
    for (int d = t; d < 512; d += 256) {
        for (int nn = 0; nn < 4; ++nn) {
            float acc = 0.0f;
            for (int mm = 0; mm < 4; ++mm)
                acc += dist[nn * 4 + mm] * vb[mm * 2048 + d];
            ob[nn * 512 + d] = acc;
        }
    }
}

// 3x3 VALID conv over att [B,4,4,512] -> relu -> c [B,4,2,510] flattened [B,4080]
__global__ void conv_relu_kernel(const float* __restrict__ att, const float* __restrict__ cw,
                                 const float* __restrict__ cb, float* __restrict__ c) {
    int idx = blockIdx.x * blockDim.x + threadIdx.x;
    const int total = BATCH * 4 * 2 * 510;
    if (idx >= total) return;
    int ow = idx % 510; int tmp = idx / 510;
    int oh = tmp % 2;   tmp /= 2;
    int co = tmp % 4;   int b = tmp / 4;
    float acc = cb[co];
    for (int ci = 0; ci < 4; ++ci)
        for (int kh = 0; kh < 3; ++kh)
            for (int kw = 0; kw < 3; ++kw)
                acc += att[((size_t)b * 4 + ci) * 2048 + (oh + kh) * 512 + (ow + kw)] *
                       cw[((co * 4 + ci) * 3 + kh) * 3 + kw];
    c[(size_t)b * 4080 + co * 1020 + oh * 510 + ow] = fmaxf(acc, 0.0f);
}

// ---------------------------------------------------------------------------
extern "C" void kernel_launch(void* const* d_in, const int* in_sizes, int n_in,
                              void* d_out, int out_size, void* d_ws, size_t ws_size,
                              hipStream_t stream) {
    const float* x         = (const float*)d_in[0];
    const int*   edge      = (const int*)d_in[1];     // [2, N_EDGES]
    const int*   batch     = (const int*)d_in[2];
    const int*   hedge     = (const int*)d_in[3];     // [2, N_HNNZ]
    const float* fp        = (const float*)d_in[4];
    const float* text_em   = (const float*)d_in[5];
    const float* fc1_W  = (const float*)d_in[7],  *fc1_b  = (const float*)d_in[8];
    const float* fc2_W  = (const float*)d_in[9],  *fc2_b  = (const float*)d_in[10];
    const float* fc3_W  = (const float*)d_in[11], *fc3_b  = (const float*)d_in[12];
    const float* fcg1_W = (const float*)d_in[13], *fcg1_b = (const float*)d_in[14];
    const float* fcg2_W = (const float*)d_in[15], *fcg2_b = (const float*)d_in[16];
    const float* fchg1_W = (const float*)d_in[17], *fchg1_b = (const float*)d_in[18];
    const float* fchg2_W = (const float*)d_in[19], *fchg2_b = (const float*)d_in[20];
    const float* gin1_W = (const float*)d_in[21], *gin1_b = (const float*)d_in[22];
    const float* gin2_W = (const float*)d_in[23], *gin2_b = (const float*)d_in[24];
    const float* h3_W   = (const float*)d_in[25], *h3_b   = (const float*)d_in[26];
    const float* h4_W   = (const float*)d_in[27], *h4_b   = (const float*)d_in[28];
    const float* Wq     = (const float*)d_in[29];
    const float* Wk     = (const float*)d_in[30];
    const float* Wv     = (const float*)d_in[31];
    const float* conv_W = (const float*)d_in[32], *conv_b = (const float*)d_in[33];
    const float* mlp1_W = (const float*)d_in[34], *mlp1_b = (const float*)d_in[35];
    const float* mlp2_W = (const float*)d_in[36], *mlp2_b = (const float*)d_in[37];

    float* out = (float*)d_out;                     // [0:1024) out, then x_g, then hx_g
    float* xg_out  = out + 1024;
    float* hxg_out = out + 1024 + 1024 * 512;

    // ---- workspace layout (floats) ----
    float* ws = (float*)d_ws;
    size_t off = 0;
    auto alloc = [&](size_t n) { float* p = ws + off; off += n; return p; };
    float* AGG  = alloc((size_t)N_NODES * 84);     // gin aggregation; later attn output (with XG1)
    float* XG1  = alloc((size_t)N_NODES * 84);     // gin hidden (contiguous after AGG)
    float* BIG1 = alloc((size_t)N_NODES * 840);    // xg2; later hyper nodes; later q/k/v
    float* EH   = alloc((size_t)N_HEDGES * 256);   // hyperedge accumulators
    float* POOL = alloc((size_t)BATCH * 1680);
    float* H1   = alloc((size_t)BATCH * 1024);
    float* FPH  = alloc((size_t)BATCH * 128);
    float* INT_ = alloc((size_t)BATCH * 2048);     // in_t [B,4,512]
    float* CBUF = alloc((size_t)BATCH * 4080);
    float* BINV = alloc(N_HEDGES);
    float* DINV = alloc(N_NODES);
    int*   STARTS = (int*)alloc(BATCH + 1);

    const int* e_src = edge;                 // row 0
    const int* e_dst = edge + N_EDGES;       // row 1
    const int* nid   = hedge;                // row 0
    const int* eid   = hedge + N_HNNZ;       // row 1

    auto gemm = [&](const float* A, int lda, const float* Wt, const float* bias,
                    float* C, int ldc, int M, int N, int K, int act) {
        dim3 g((M + BM - 1) / BM, (N + BN - 1) / BN);
        wmma_gemm_kernel<<<g, 256, 0, stream>>>(A, lda, Wt, bias, C, ldc, M, N, K, act);
    };
    auto zero = [&](float* p, int n) {
        zero_kernel<<<(n + 255) / 256, 256, 0, stream>>>(p, n);
    };

    // ===== fingerprint MLP -> in_t token 0 =====
    gemm(fp, 2513, fc1_W, fc1_b, FPH, 128, BATCH, 128, 2513, 1);
    gemm(FPH, 128, fc2_W, fc2_b, INT_ + 0, 2048, BATCH, 512, 128, 0);

    // ===== text -> in_t token 3 =====
    gemm(text_em, 768, fc3_W, fc3_b, INT_ + 3 * 512, 2048, BATCH, 512, 768, 0);

    // ===== GIN branch =====
    zero(AGG, N_NODES * 84);
    scatter_add_kernel<<<((size_t)N_EDGES * 84 + 255) / 256, 256, 0, stream>>>(
        AGG, e_dst, x, e_src, N_EDGES, 84);
    add_inplace_kernel<<<(N_NODES * 84 + 255) / 256, 256, 0, stream>>>(AGG, x, N_NODES * 84);
    gemm(AGG, 84, gin1_W, gin1_b, XG1, 84, N_NODES, 84, 84, 1);

    zero(AGG, N_NODES * 84);
    scatter_add_kernel<<<((size_t)N_EDGES * 84 + 255) / 256, 256, 0, stream>>>(
        AGG, e_dst, XG1, e_src, N_EDGES, 84);
    add_inplace_kernel<<<(N_NODES * 84 + 255) / 256, 256, 0, stream>>>(AGG, XG1, N_NODES * 84);
    gemm(AGG, 84, gin2_W, gin2_b, BIG1, 840, N_NODES, 840, 84, 1);   // xg2 [N,840]

    starts_kernel<<<(N_NODES + 255) / 256, 256, 0, stream>>>(STARTS, batch, N_NODES, BATCH);
    {
        dim3 g(BATCH, (840 + 255) / 256);
        pool_kernel<<<g, 256, 0, stream>>>(BIG1, 840, STARTS, POOL);   // [B,1680]
    }
    gemm(POOL, 1680, fcg1_W, fcg1_b, H1, 1024, BATCH, 1024, 1680, 1);
    gemm(H1, 1024, fcg2_W, fcg2_b, INT_ + 512, 2048, BATCH, 512, 1024, 0);   // x_g token 1
    copy_strided_kernel<<<(BATCH * 512 + 255) / 256, 256, 0, stream>>>(
        xg_out, 512, INT_ + 512, 2048, BATCH, 512);

    // ===== Hypergraph branch =====
    zero(BINV, N_HEDGES);
    count_kernel<<<(N_HNNZ + 255) / 256, 256, 0, stream>>>(BINV, eid, N_HNNZ);
    binv_kernel<<<(N_HEDGES + 255) / 256, 256, 0, stream>>>(BINV, N_HEDGES);
    zero(DINV, N_NODES);
    count_kernel<<<(N_HNNZ + 255) / 256, 256, 0, stream>>>(DINV, nid, N_HNNZ);
    dinv_kernel<<<(N_NODES + 255) / 256, 256, 0, stream>>>(DINV, N_NODES);

    // layer 1: 84 -> 168   (xw1 in AGG+XG1 contiguous region, hnode1 at BIG1)
    float* XW1 = AGG;                              // N*168 spans AGG..XG1
    float* HN1 = BIG1;                             // N*168
    float* XW2 = BIG1 + (size_t)N_NODES * 168;     // N*256
    float* HN2 = BIG1 + (size_t)N_NODES * 424;     // N*256
    gemm(x, 84, h3_W, nullptr, XW1, 168, N_NODES, 168, 84, 0);
    zero(EH, N_HEDGES * 168);
    scatter_add_kernel<<<((size_t)N_HNNZ * 168 + 255) / 256, 256, 0, stream>>>(
        EH, eid, XW1, nid, N_HNNZ, 168);
    scale_rows_kernel<<<(N_HEDGES * 168 + 255) / 256, 256, 0, stream>>>(EH, BINV, N_HEDGES, 168);
    zero(HN1, N_NODES * 168);
    scatter_add_kernel<<<((size_t)N_HNNZ * 168 + 255) / 256, 256, 0, stream>>>(
        HN1, nid, EH, eid, N_HNNZ, 168);
    scale_bias_rows_kernel<<<(N_NODES * 168 + 255) / 256, 256, 0, stream>>>(
        HN1, DINV, h3_b, N_NODES, 168);

    // layer 2: 168 -> 256
    gemm(HN1, 168, h4_W, nullptr, XW2, 256, N_NODES, 256, 168, 0);
    zero(EH, N_HEDGES * 256);
    scatter_add_kernel<<<((size_t)N_HNNZ * 256 + 255) / 256, 256, 0, stream>>>(
        EH, eid, XW2, nid, N_HNNZ, 256);
    scale_rows_kernel<<<(N_HEDGES * 256 + 255) / 256, 256, 0, stream>>>(EH, BINV, N_HEDGES, 256);
    zero(HN2, N_NODES * 256);
    scatter_add_kernel<<<((size_t)N_HNNZ * 256 + 255) / 256, 256, 0, stream>>>(
        HN2, nid, EH, eid, N_HNNZ, 256);
    scale_bias_rows_kernel<<<(N_NODES * 256 + 255) / 256, 256, 0, stream>>>(
        HN2, DINV, h4_b, N_NODES, 256);

    {
        dim3 g(BATCH, (256 + 255) / 256);
        pool_kernel<<<g, 256, 0, stream>>>(HN2, 256, STARTS, POOL);    // [B,512]
    }
    gemm(POOL, 512, fchg1_W, fchg1_b, H1, 1024, BATCH, 1024, 512, 1);
    gemm(H1, 1024, fchg2_W, fchg2_b, INT_ + 1024, 2048, BATCH, 512, 1024, 0);  // hx_g token 2
    copy_strided_kernel<<<(BATCH * 512 + 255) / 256, 256, 0, stream>>>(
        hxg_out, 512, INT_ + 1024, 2048, BATCH, 512);

    // ===== fusion attention =====
    float* Q = BIG1;
    float* K_ = BIG1 + (size_t)4096 * 2048;
    float* V = BIG1 + (size_t)2 * 4096 * 2048;
    gemm(INT_, 512, Wq, nullptr, Q,  2048, 4096, 2048, 512, 0);
    gemm(INT_, 512, Wk, nullptr, K_, 2048, 4096, 2048, 512, 0);
    gemm(INT_, 512, Wv, nullptr, V,  2048, 4096, 2048, 512, 0);

    float* ATT = AGG;  // AGG+XG1 contiguous: N*168 floats >= B*4*4*512
    attn_kernel<<<BATCH * 4, 256, 0, stream>>>(Q, K_, V, ATT);

    conv_relu_kernel<<<(BATCH * 4 * 2 * 510 + 255) / 256, 256, 0, stream>>>(
        ATT, conv_W, conv_b, CBUF);

    // ===== output MLP =====
    gemm(CBUF, 4080, mlp1_W, mlp1_b, H1, 1024, BATCH, 1024, 4080, 1);
    gemm(H1, 1024, mlp2_W, mlp2_b, out, 1, BATCH, 1, 1024, 0);

    (void)in_sizes; (void)n_in; (void)out_size; (void)ws_size;
}
